// MoELayer_84310208020547
// MI455X (gfx1250) — matrix operand order repaired and checked by reading, not compiled, not run on hardware
//
#include <hip/hip_runtime.h>
#include <hip/hip_bf16.h>
#include <math.h>

#define DIMSZ 1024
#define NEXP  8
#define TOPK  2
#define HID   4096
#define NTOK  8192

#define BM 128
#define BN 128
#define BK 32
#define LSTR 40   // LDS row stride in bf16 elems (32 data + 8 pad), 80B keeps 16B alignment

typedef __attribute__((ext_vector_type(16))) __bf16 v16bf;
typedef __attribute__((ext_vector_type(8)))  __bf16 v8bf;
typedef __attribute__((ext_vector_type(8)))  float  v8f;
typedef __attribute__((ext_vector_type(4)))  unsigned int u32x4;
typedef __attribute__((ext_vector_type(8)))  int          i32x8;
typedef __attribute__((ext_vector_type(4)))  int          i32x4;

union FragBF { v16bf v; v8bf h[2]; };

#if __has_builtin(__builtin_amdgcn_tensor_load_to_lds) && __has_builtin(__builtin_amdgcn_s_wait_tensorcnt)
#define HAVE_TDM 1
#else
#define HAVE_TDM 0
#endif

__device__ __forceinline__ unsigned short f2bf(float f) {
  unsigned int u = __float_as_uint(f);
  u += 0x7FFFu + ((u >> 16) & 1u);           // round-to-nearest-even
  return (unsigned short)(u >> 16);
}
__device__ __forceinline__ unsigned int pack2(float lo, float hi) {
  return (unsigned int)f2bf(lo) | ((unsigned int)f2bf(hi) << 16);
}
__device__ __forceinline__ float gelu_exact(float v) {
  return 0.5f * v * (1.0f + erff(v * 0.70710678118654752f));
}

struct R16 { float4 f[4]; };   // 16 fp32 staged in regs

// 16 contiguous fp32 (A tile row chunk)
__device__ __forceinline__ R16 ldg16(const float* __restrict__ p) {
  R16 r;
  r.f[0] = *(const float4*)(p + 0);
  r.f[1] = *(const float4*)(p + 4);
  r.f[2] = *(const float4*)(p + 8);
  r.f[3] = *(const float4*)(p + 12);
  return r;
}
// 8 fp32 from each of two consecutive K rows (B tile)
__device__ __forceinline__ R16 ldgB(const float* __restrict__ p, size_t rowStride) {
  R16 r;
  r.f[0] = *(const float4*)(p + 0);
  r.f[1] = *(const float4*)(p + 4);
  r.f[2] = *(const float4*)(p + rowStride + 0);
  r.f[3] = *(const float4*)(p + rowStride + 4);
  return r;
}
// A tile: 16 contiguous bf16, packed -> two b128 LDS stores; branchless zeroing
__device__ __forceinline__ void stA(unsigned short* d, const R16& r, bool valid) {
  float v[16];
  *(float4*)(v + 0) = r.f[0]; *(float4*)(v + 4) = r.f[1];
  *(float4*)(v + 8) = r.f[2]; *(float4*)(v + 12) = r.f[3];
  unsigned int u[8];
#pragma unroll
  for (int i = 0; i < 8; i++) {
    float a = valid ? v[2 * i]     : 0.f;
    float b = valid ? v[2 * i + 1] : 0.f;
    u[i] = pack2(a, b);
  }
  uint4* q = (uint4*)d;
  q[0] = make_uint4(u[0], u[1], u[2], u[3]);
  q[1] = make_uint4(u[4], u[5], u[6], u[7]);
}
// B tile (transposed [n][k]): 8 columns x one k-pair -> 8 packed b32 LDS stores
__device__ __forceinline__ void stB(unsigned short* lB, int n0, int k2, const R16& r) {
  float a[8], b[8];
  *(float4*)(a + 0) = r.f[0]; *(float4*)(a + 4) = r.f[1];
  *(float4*)(b + 0) = r.f[2]; *(float4*)(b + 4) = r.f[3];
#pragma unroll
  for (int i = 0; i < 8; i++)
    *(unsigned int*)&lB[(n0 + i) * LSTR + k2] = pack2(a[i], b[i]);
}

#if HAVE_TDM
// TDM: load a BM x BK bf16 tile (row stride 4096 elems) into LDS with
// 16B pad after each 64B row (reproduces LSTR=40 layout). One op per block.
__device__ __forceinline__ void tdm_load_A(unsigned lds_off_bytes,
                                           const unsigned short* gaddr) {
  unsigned long long ga = (unsigned long long)(size_t)gaddr;
  u32x4 g0;
  g0[0] = 1u;                                           // count=1, user D#
  g0[1] = lds_off_bytes;                                // lds_addr
  g0[2] = (unsigned)(ga & 0xFFFFFFFFu);                 // global_addr lo
  g0[3] = (unsigned)((ga >> 32) & 0x01FFFFFFu) | 0x80000000u; // addr hi | type=2
  i32x8 g1;
  g1[0] = (int)((1u << 16) |      // data_size = 2B
                (1u << 20) |      // pad_enable
                (3u << 22) |      // pad_interval: 16 dwords (64B row)
                (3u << 25));      // pad_amount: 4 dwords (16B pad)
  g1[1] = (int)((unsigned)HID << 16);   // tensor_dim0 = 4096 (lo16 -> [31:16])
  g1[2] = (int)((unsigned)BM << 16);    // tensor_dim0 hi=0 | tensor_dim1 lo16 = 128
  g1[3] = (int)((unsigned)BK << 16);    // tensor_dim1 hi=0 | tile_dim0 = 32
  g1[4] = BM;                           // tile_dim1 = 128, tile_dim2 = 0
  g1[5] = HID;                          // tensor_dim0_stride lo = 4096
  g1[6] = 0;
  g1[7] = 0;
  i32x4 z4 = {0, 0, 0, 0};
  i32x8 z8 = {0, 0, 0, 0, 0, 0, 0, 0};
  __builtin_amdgcn_tensor_load_to_lds(g0, g1, z4, z4, z8, 0);
}
#endif

// ---------------- init: zero output + counters ----------------
__global__ __launch_bounds__(256) void init_kernel(float* __restrict__ out,
                                                   int* __restrict__ counts,
                                                   int* __restrict__ cursor) {
  size_t i  = (size_t)blockIdx.x * blockDim.x + threadIdx.x;
  size_t n4 = (size_t)NTOK * DIMSZ / 4;
  float4 z = make_float4(0.f, 0.f, 0.f, 0.f);
  for (size_t j = i; j < n4; j += (size_t)gridDim.x * blockDim.x)
    ((float4*)out)[j] = z;
  if (blockIdx.x == 0 && threadIdx.x < NEXP) { counts[threadIdx.x] = 0; cursor[threadIdx.x] = 0; }
}

// ---------------- router: one wave32 per token ----------------
__global__ __launch_bounds__(256) void router_kernel(const float* __restrict__ x,
                                                     const float* __restrict__ Wr,
                                                     int* __restrict__ counts,
                                                     int* __restrict__ eIdx,
                                                     float* __restrict__ eW) {
  const int lane = threadIdx.x & 31, wv = threadIdx.x >> 5;
  const int t = blockIdx.x * 8 + wv;
  if (t >= NTOK) return;
  const float* __restrict__ xr = x + (size_t)t * DIMSZ;
  float acc[NEXP];
#pragma unroll
  for (int e = 0; e < NEXP; e++) acc[e] = 0.f;
  for (int i = lane; i < DIMSZ; i += 32) {
    float xv = xr[i];
    const float* __restrict__ wr = Wr + (size_t)i * NEXP;
#pragma unroll
    for (int e = 0; e < NEXP; e++) acc[e] += xv * wr[e];
  }
#pragma unroll
  for (int off = 16; off > 0; off >>= 1) {
#pragma unroll
    for (int e = 0; e < NEXP; e++) acc[e] += __shfl_xor(acc[e], off, 32);
  }
  if (lane == 0) {
    int i0 = 0;
#pragma unroll
    for (int e = 1; e < NEXP; e++) if (acc[e] > acc[i0]) i0 = e;
    int i1 = (i0 == 0) ? 1 : 0;
#pragma unroll
    for (int e = 0; e < NEXP; e++) if (e != i0 && acc[e] > acc[i1]) i1 = e;
    float p1  = expf(acc[i1] - acc[i0]);     // softmax over top-2
    float inv = 1.0f / (1.0f + p1);
    eIdx[t * 2 + 0] = i0; eIdx[t * 2 + 1] = i1;
    eW[t * 2 + 0] = inv;  eW[t * 2 + 1] = p1 * inv;
    atomicAdd(&counts[i0], 1);
    atomicAdd(&counts[i1], 1);
  }
}

// ---------------- scan: aligned exclusive scan over 8 experts ----------------
__global__ void scan_kernel(const int* __restrict__ counts, int* __restrict__ offA) {
  if (threadIdx.x == 0 && blockIdx.x == 0) {
    int off = 0;
    for (int e = 0; e < NEXP; e++) { offA[e] = off; off += (counts[e] + BM - 1) & ~(BM - 1); }
    offA[NEXP] = off;
  }
}

// ---------------- bucket: compact (token, weight) per expert ----------------
__global__ __launch_bounds__(256) void bucket_kernel(const int* __restrict__ eIdx,
                                                     const float* __restrict__ eW,
                                                     const int* __restrict__ offA,
                                                     int* __restrict__ cursor,
                                                     int* __restrict__ tok_id,
                                                     float* __restrict__ tok_w) {
  const int t = blockIdx.x * 256 + threadIdx.x;
  if (t >= NTOK) return;
#pragma unroll
  for (int j = 0; j < TOPK; j++) {
    int e = eIdx[t * 2 + j];
    int slot = atomicAdd(&cursor[e], 1);
    int p = offA[e] + slot;
    tok_id[p] = t;
    tok_w[p]  = eW[t * 2 + j];
  }
}

// ---------------- GEMM1: H = gelu(Xg @ W1[e] + b1[e]) in bf16 ----------------
__global__ __launch_bounds__(256) void gemm1_kernel(const float* __restrict__ x,
                                                    const float* __restrict__ W1,
                                                    const float* __restrict__ b1,
                                                    const int* __restrict__ tok_id,
                                                    const int* __restrict__ offA,
                                                    const int* __restrict__ counts,
                                                    unsigned short* __restrict__ Hbuf) {
  const int e = blockIdx.z;
  const int cnt = counts[e];
  const int cntPad = (cnt + BM - 1) & ~(BM - 1);
  const int rowBase = blockIdx.y * BM;
  if (rowBase >= cntPad) return;
  const int colBase = blockIdx.x * BN;
  const int base = offA[e];
  const float* __restrict__ W = W1 + (size_t)e * DIMSZ * HID;

  __shared__ unsigned short lA0[BM * LSTR], lA1[BM * LSTR];
  __shared__ unsigned short lB0[BN * LSTR], lB1[BN * LSTR];

  const int tid = threadIdx.x;
  const int lane = tid & 31, wv = tid >> 5;
  const int wm = (wv & 1) * 64, wn = (wv >> 1) * 32;   // 2x4 waves -> 64x32 each
  const int halfA = lane >> 4, lr = lane & 15;

  // A gather addressing: thread -> (row, 16-wide K chunk); clamp to a safe row
  const int ar = tid >> 1;
  const int ak = (tid & 1) * 16;
  const int tokRow = rowBase + ar;
  const bool avalid = tokRow < cnt;                 // cnt >= 1 here
  const int tok = tok_id[base + (avalid ? tokRow : 0)];
  const float* __restrict__ xrow = x + (size_t)tok * DIMSZ;

  // B staging addressing: thread -> (k-pair, 8-wide N chunk), transpose into LDS
  const int k2  = (tid >> 4) * 2;       // 0,2,...,30
  const int bn0 = (tid & 15) * 8;       // 0..120

  v8f acc[4][2];
#pragma unroll
  for (int m = 0; m < 4; m++)
#pragma unroll
    for (int n = 0; n < 2; n++)
#pragma unroll
      for (int r = 0; r < 8; r++) acc[m][n][r] = 0.f;

  // prologue: stage tile 0 into buffer 0
  {
    R16 a = ldg16(xrow + ak);
    R16 b = ldgB(W + (size_t)k2 * HID + colBase + bn0, HID);
    stA(&lA0[ar * LSTR + ak], a, avalid);
    stB(lB0, bn0, k2, b);
  }
  __syncthreads();

  unsigned short* pAc = lA0; unsigned short* pAn = lA1;
  unsigned short* pBc = lB0; unsigned short* pBn = lB1;

  for (int k0 = 0; k0 < DIMSZ; k0 += BK) {
    const bool more = (k0 + BK) < DIMSZ;
    R16 nA, nB;
    if (more) {   // issue next-tile global loads; waits land before the stores below
      nA = ldg16(xrow + k0 + BK + ak);
      nB = ldgB(W + (size_t)(k0 + BK + k2) * HID + colBase + bn0, HID);
    }

    FragBF af[4], bfr[2];
#pragma unroll
    for (int m = 0; m < 4; m++) {
      const unsigned short* rp = pAc + (wm + m * 16 + lr) * LSTR;
      af[m].h[0] = *(const v8bf*)(rp + halfA * 8);
      af[m].h[1] = *(const v8bf*)(rp + halfA * 8 + 16);
    }
#pragma unroll
    for (int n = 0; n < 2; n++) {
      const unsigned short* rp = pBc + (wn + n * 16 + lr) * LSTR;
      bfr[n].h[0] = *(const v8bf*)(rp + halfA * 8);
      bfr[n].h[1] = *(const v8bf*)(rp + halfA * 8 + 16);
    }
#pragma unroll
    for (int m = 0; m < 4; m++)
#pragma unroll
      for (int n = 0; n < 2; n++)
        acc[m][n] = __builtin_amdgcn_wmma_f32_16x16x32_bf16(
            false, af[m].v, false, bfr[n].v, (short)0, acc[m][n], false, false);

    if (more) {
      stA(&pAn[ar * LSTR + ak], nA, avalid);
      stB(pBn, bn0, k2, nB);
    }
    __syncthreads();
    unsigned short* t0 = pAc; pAc = pAn; pAn = t0;
    unsigned short* t1 = pBc; pBc = pBn; pBn = t1;
  }

  // epilogue: bias + exact GELU -> bf16 H
  const size_t hbase = (size_t)(base + rowBase) * HID;
#pragma unroll
  for (int m = 0; m < 4; m++) {
    const int mrow = wm + m * 16 + halfA * 8;
#pragma unroll
    for (int n = 0; n < 2; n++) {
      const int gN = colBase + wn + n * 16 + lr;
      const float bb = b1[e * HID + gN];
#pragma unroll
      for (int r = 0; r < 8; r++) {
        float v = acc[m][n][r] + bb;
        Hbuf[hbase + (size_t)(mrow + r) * HID + gN] = f2bf(gelu_exact(v));
      }
    }
  }
}

// ---------------- GEMM2: out += w * (H @ W2[e] + b2[e]) ----------------
__global__ __launch_bounds__(256) void gemm2_kernel(const unsigned short* __restrict__ Hbuf,
                                                    const float* __restrict__ W2,
                                                    const float* __restrict__ b2,
                                                    const int* __restrict__ tok_id,
                                                    const float* __restrict__ tok_w,
                                                    const int* __restrict__ offA,
                                                    const int* __restrict__ counts,
                                                    float* __restrict__ out) {
  const int e = blockIdx.z;
  const int cnt = counts[e];
  const int cntPad = (cnt + BM - 1) & ~(BM - 1);
  const int rowBase = blockIdx.y * BM;
  if (rowBase >= cntPad) return;
  const int colBase = blockIdx.x * BN;
  const int base = offA[e];
  const float* __restrict__ W = W2 + (size_t)e * HID * DIMSZ;

  __shared__ unsigned short lA0[BM * LSTR], lA1[BM * LSTR];
  __shared__ unsigned short lB0[BN * LSTR], lB1[BN * LSTR];

  const int tid = threadIdx.x;
  const int lane = tid & 31, wv = tid >> 5;
  const int wm = (wv & 1) * 64, wn = (wv >> 1) * 32;
  const int halfA = lane >> 4, lr = lane & 15;

  const int ar = tid >> 1;
  const int ak = (tid & 1) * 16;
  // tile base of gathered H rows for this block (rows rowBase..rowBase+127 all exist in Hbuf)
  const unsigned short* __restrict__ hTile = Hbuf + (size_t)(base + rowBase) * HID;
  const unsigned short* __restrict__ hrow  = hTile + (size_t)ar * HID;
  const int k2  = (tid >> 4) * 2;
  const int bn0 = (tid & 15) * 8;

#if HAVE_TDM
  const unsigned ldsA0 = (unsigned)(size_t)&lA0[0];   // low 32 bits of generic ptr = LDS offset
  const unsigned ldsA1 = (unsigned)(size_t)&lA1[0];
#endif

  v8f acc[4][2];
#pragma unroll
  for (int m = 0; m < 4; m++)
#pragma unroll
    for (int n = 0; n < 2; n++)
#pragma unroll
      for (int r = 0; r < 8; r++) acc[m][n][r] = 0.f;

  // prologue: stage tile 0
#if HAVE_TDM
  if (wv == 0) tdm_load_A(ldsA0, hTile);
#else
  {
    const uint4* s = (const uint4*)(hrow + ak);
    uint4* d = (uint4*)&lA0[ar * LSTR + ak];
    d[0] = s[0]; d[1] = s[1];
  }
#endif
  {
    R16 b = ldgB(W + (size_t)k2 * DIMSZ + colBase + bn0, DIMSZ);
    stB(lB0, bn0, k2, b);
  }
#if HAVE_TDM
  if (wv == 0) __builtin_amdgcn_s_wait_tensorcnt(0);
#endif
  __syncthreads();

  unsigned short* pAc = lA0; unsigned short* pAn = lA1;
  unsigned short* pBc = lB0; unsigned short* pBn = lB1;
#if HAVE_TDM
  unsigned ldsCur = ldsA0, ldsNxt = ldsA1;
#endif

  for (int k0 = 0; k0 < HID; k0 += BK) {
    const bool more = (k0 + BK) < HID;
    R16 nB;
    if (more) {
#if HAVE_TDM
      if (wv == 0) tdm_load_A(ldsNxt, hTile + k0 + BK);
#endif
      nB = ldgB(W + (size_t)(k0 + BK + k2) * DIMSZ + colBase + bn0, DIMSZ);
    }
#if !HAVE_TDM
    uint4 nA0, nA1;
    if (more) {
      const uint4* s = (const uint4*)(hrow + k0 + BK + ak);
      nA0 = s[0]; nA1 = s[1];
    }
#endif

    FragBF af[4], bfr[2];
#pragma unroll
    for (int m = 0; m < 4; m++) {
      const unsigned short* rp = pAc + (wm + m * 16 + lr) * LSTR;
      af[m].h[0] = *(const v8bf*)(rp + halfA * 8);
      af[m].h[1] = *(const v8bf*)(rp + halfA * 8 + 16);
    }
#pragma unroll
    for (int n = 0; n < 2; n++) {
      const unsigned short* rp = pBc + (wn + n * 16 + lr) * LSTR;
      bfr[n].h[0] = *(const v8bf*)(rp + halfA * 8);
      bfr[n].h[1] = *(const v8bf*)(rp + halfA * 8 + 16);
    }
#pragma unroll
    for (int m = 0; m < 4; m++)
#pragma unroll
      for (int n = 0; n < 2; n++)
        acc[m][n] = __builtin_amdgcn_wmma_f32_16x16x32_bf16(
            false, af[m].v, false, bfr[n].v, (short)0, acc[m][n], false, false);

    if (more) {
      stB(pBn, bn0, k2, nB);
#if !HAVE_TDM
      uint4* d = (uint4*)&pAn[ar * LSTR + ak];
      d[0] = nA0; d[1] = nA1;
#endif
#if HAVE_TDM
      if (wv == 0) __builtin_amdgcn_s_wait_tensorcnt(0);
#endif
    }
    __syncthreads();
    unsigned short* t0 = pAc; pAc = pAn; pAn = t0;
    unsigned short* t1 = pBc; pBc = pBn; pBn = t1;
#if HAVE_TDM
    unsigned tl = ldsCur; ldsCur = ldsNxt; ldsNxt = tl;
#endif
  }

  // epilogue: bias, scale by routing weight, scatter-accumulate
#pragma unroll
  for (int m = 0; m < 4; m++) {
    const int mrow = wm + m * 16 + halfA * 8;
#pragma unroll
    for (int r = 0; r < 8; r++) {
      const int gM = rowBase + mrow + r;
      if (gM < cnt) {
        const int t = tok_id[base + gM];
        const float w = tok_w[base + gM];
        float* __restrict__ orow = out + (size_t)t * DIMSZ;
#pragma unroll
        for (int n = 0; n < 2; n++) {
          const int gN = colBase + wn + n * 16 + lr;
          float v = acc[m][n][r] + b2[e * DIMSZ + gN];
          atomicAdd(&orow[gN], w * v);
        }
      }
    }
  }
}

// ---------------- launch ----------------
extern "C" void kernel_launch(void* const* d_in, const int* in_sizes, int n_in,
                              void* d_out, int out_size, void* d_ws, size_t ws_size,
                              hipStream_t stream) {
  const float* x  = (const float*)d_in[0];
  const float* Wr = (const float*)d_in[1];
  const float* W1 = (const float*)d_in[2];
  const float* b1 = (const float*)d_in[3];
  const float* W2 = (const float*)d_in[4];
  const float* b2 = (const float*)d_in[5];
  float* out = (float*)d_out;
  (void)in_sizes; (void)n_in; (void)out_size; (void)ws_size;

  char* ws = (char*)d_ws;
  size_t off = 0;
  auto alloc = [&](size_t bytes) -> char* {
    char* p = ws + off;
    off = (off + bytes + 255) & ~(size_t)255;
    return p;
  };
  const size_t maxRows = (size_t)NTOK * TOPK + (size_t)NEXP * BM; // padded gathered rows
  int*   counts = (int*)alloc(NEXP * sizeof(int));
  int*   cursor = (int*)alloc(NEXP * sizeof(int));
  int*   offA   = (int*)alloc((NEXP + 1) * sizeof(int));
  int*   eIdx   = (int*)alloc((size_t)NTOK * 2 * sizeof(int));
  float* eW     = (float*)alloc((size_t)NTOK * 2 * sizeof(float));
  int*   tok_id = (int*)alloc(maxRows * sizeof(int));
  float* tok_w  = (float*)alloc(maxRows * sizeof(float));
  unsigned short* Hbuf = (unsigned short*)alloc(maxRows * HID * sizeof(unsigned short));

  init_kernel<<<2048, 256, 0, stream>>>(out, counts, cursor);
  router_kernel<<<NTOK / 8, 256, 0, stream>>>(x, Wr, counts, eIdx, eW);
  scan_kernel<<<1, 32, 0, stream>>>(counts, offA);
  bucket_kernel<<<NTOK / 256, 256, 0, stream>>>(eIdx, eW, offA, cursor, tok_id, tok_w);

  dim3 g1(HID / BN, NTOK / BM, NEXP);   // 32 x 64 x 8, most row-blocks early-exit
  gemm1_kernel<<<g1, 256, 0, stream>>>(x, W1, b1, tok_id, offA, counts, Hbuf);

  dim3 g2(DIMSZ / BN, NTOK / BM, NEXP); // 8 x 64 x 8
  gemm2_kernel<<<g2, 256, 0, stream>>>(Hbuf, W2, b2, tok_id, tok_w, offA, counts, out);
}
